// SDSA_19894288515537
// MI455X (gfx1250) — matrix operand order repaired and checked by reading, not compiled, not run on hardware
//
#include <hip/hip_runtime.h>
#include <hip/hip_bf16.h>

// ---------------------------------------------------------------------------
// SDSA (spiking self-attention) for MI455X / gfx1250.
// T=4, S=2048, D=768, H=12, hd=64.
// All heavy math (4x [8192x768]@[768x768]) runs on v_wmma_f32_16x16x32_bf16.
// ---------------------------------------------------------------------------

typedef __bf16  bf16_t;
typedef __attribute__((ext_vector_type(16))) __bf16 v16bf;
typedef __attribute__((ext_vector_type(8)))  float  v8f;

#define TT 4
#define SS 2048
#define DD 768
#define SD (SS * DD)          // 1,572,864
#define TSD (TT * SD)         // 6,291,456
#define NH 12
#define HD 64

// ---------------------------------------------------------------------------
// 1) Weight convert f32 (K,N) -> bf16 transposed (N,K)
// ---------------------------------------------------------------------------
__global__ void convert_w_kernel(const float* __restrict__ w, bf16_t* __restrict__ wt) {
  int idx = blockIdx.x * 256 + threadIdx.x;       // over 768*768, output index (n,k)
  if (idx >= DD * DD) return;
  int n = idx / DD;
  int k = idx - n * DD;
  wt[idx] = (bf16_t)w[(size_t)k * DD + n];
}

// ---------------------------------------------------------------------------
// 2) IF node on x -> bf16 spikes (exact: spikes are {0,1})
// ---------------------------------------------------------------------------
__global__ void ifnode0_kernel(const float* __restrict__ x, bf16_t* __restrict__ spk) {
  int idx = blockIdx.x * 256 + threadIdx.x;       // over S*D
  if (idx >= SD) return;
  float v = 0.f;
  for (int t = 0; t < TT; ++t) {
    size_t o = (size_t)t * SD + idx;
    v += x[o];
    float s = (v >= 1.f) ? 1.f : 0.f;             // heaviside(v - 1)
    spk[o] = (bf16_t)s;
    v = (s != 0.f) ? 0.f : v;                     // hard reset
  }
}

// ---------------------------------------------------------------------------
// 3) Tiled bf16 WMMA GEMM:  C[M,N] (f32) = A[M,K] (bf16) @ Bt[N,K]^T (bf16)
//    256 threads = 8 waves. 128x128 tile, K-step 32.
//    Wave (wm in 0..3, wn in 0..1) owns rows wm*32..+31, cols wn*64..+63
//    => 2x4 grid of 16x16 wmma accumulators per wave.
// ---------------------------------------------------------------------------
#define TM 128
#define TN 128
#define TK 32
#define LDSP (TK + 8)   // 40 bf16 = 80B row stride: 16B-aligned, conflict-poor

__global__ __launch_bounds__(256, 2)
void gemm_bf16_wmma_kernel(const bf16_t* __restrict__ A,
                           const bf16_t* __restrict__ Bt,
                           float* __restrict__ C,
                           int M, int N, int K) {
  __shared__ __align__(16) bf16_t lds_a[TM][LDSP];
  __shared__ __align__(16) bf16_t lds_b[TN][LDSP];

  const int tid  = threadIdx.x;
  const int lane = tid & 31;
  const int wid  = tid >> 5;
  const int wm   = wid & 3;        // 32-row band
  const int wn   = wid >> 2;       // 64-col band
  const int m0   = blockIdx.x * TM;
  const int n0   = blockIdx.y * TN;

  v8f acc[2][4] = {};

  // cooperative loaders: thread t -> row t/2, 16 bf16 (two 16B chunks)
  const int lrow = tid >> 1;
  const int lk   = (tid & 1) * 16;

  const int mrow = lane & 15;
  const int kbA  = (lane >> 4) * 16;   // A: half-wave K halves
  const int kbB  = (lane >> 4) * 16;   // B: half-wave K halves

  for (int k0 = 0; k0 < K; k0 += TK) {
    const bf16_t* ga = A  + (size_t)(m0 + lrow) * K + k0 + lk;
    const bf16_t* gb = Bt + (size_t)(n0 + lrow) * K + k0 + lk;
    *(uint4*)&lds_a[lrow][lk]     = *(const uint4*)(ga);
    *(uint4*)&lds_a[lrow][lk + 8] = *(const uint4*)(ga + 8);
    *(uint4*)&lds_b[lrow][lk]     = *(const uint4*)(gb);
    *(uint4*)&lds_b[lrow][lk + 8] = *(const uint4*)(gb + 8);
    __syncthreads();

    v16bf afrag[2];
    v16bf bfrag[4];
#pragma unroll
    for (int i = 0; i < 2; ++i) {
      const bf16_t* pa = &lds_a[wm * 32 + i * 16 + mrow][kbA];
      ((uint4*)&afrag[i])[0] = *(const uint4*)(pa);        // K kb..kb+7
      ((uint4*)&afrag[i])[1] = *(const uint4*)(pa + 8);    // K kb+8..kb+15
    }
#pragma unroll
    for (int j = 0; j < 4; ++j) {
      const bf16_t* pb = &lds_b[wn * 64 + j * 16 + mrow][kbB];
      ((uint4*)&bfrag[j])[0] = *(const uint4*)(pb);        // K kbB..kbB+7
      ((uint4*)&bfrag[j])[1] = *(const uint4*)(pb + 8);    // K kbB+8..kbB+15
    }
#pragma unroll
    for (int i = 0; i < 2; ++i)
#pragma unroll
      for (int j = 0; j < 4; ++j)
        acc[i][j] = __builtin_amdgcn_wmma_f32_16x16x32_bf16(
            false, afrag[i], false, bfrag[j], (short)0, acc[i][j], false, false);
    __syncthreads();
  }

  // C/D layout: VGPR r, lanes 0-15: M=r N=lane; lanes 16-31: M=r+8 N=lane-16
  const int hi = (lane >> 4) << 3;
#pragma unroll
  for (int i = 0; i < 2; ++i) {
    const int rbase = m0 + wm * 32 + i * 16 + hi;
#pragma unroll
    for (int j = 0; j < 4; ++j) {
      const int col = n0 + wn * 64 + j * 16 + (lane & 15);
#pragma unroll
      for (int r = 0; r < 8; ++r)
        C[(size_t)(rbase + r) * N + col] = acc[i][j][r];
    }
  }
}

// ---------------------------------------------------------------------------
// 4) BN stats per channel s: mean/var over (T, D) = 3072 elems; block per s
// ---------------------------------------------------------------------------
__global__ void bn_stats_kernel(const float* __restrict__ X,
                                float* __restrict__ mean, float* __restrict__ rstd) {
  const int s = blockIdx.x;
  float sum = 0.f, sq = 0.f;
  for (int t = 0; t < TT; ++t) {
    const float* p = X + ((size_t)t * SS + s) * DD;
    for (int d = threadIdx.x; d < DD; d += 256) {
      float v = p[d];
      sum += v; sq += v * v;
    }
  }
  __shared__ float r1[256], r2[256];
  r1[threadIdx.x] = sum; r2[threadIdx.x] = sq;
  __syncthreads();
  for (int off = 128; off > 0; off >>= 1) {
    if (threadIdx.x < off) {
      r1[threadIdx.x] += r1[threadIdx.x + off];
      r2[threadIdx.x] += r2[threadIdx.x + off];
    }
    __syncthreads();
  }
  if (threadIdx.x == 0) {
    float m = r1[0] * (1.f / 3072.f);
    float var = r2[0] * (1.f / 3072.f) - m * m;      // biased variance
    mean[s] = m;
    rstd[s] = rsqrtf(var + 1e-5f);
  }
}

// ---------------------------------------------------------------------------
// 5) BN apply + IF node, in place (spikes {0,1} stored as f32)
// ---------------------------------------------------------------------------
__global__ void bn_ifnode_kernel(float* __restrict__ X,
                                 const float* __restrict__ mean,
                                 const float* __restrict__ rstd) {
  int idx = blockIdx.x * 256 + threadIdx.x;        // over S*D
  if (idx >= SD) return;
  int s = idx / DD;
  float m = mean[s], r = rstd[s];
  float v = 0.f;
  for (int t = 0; t < TT; ++t) {
    size_t o = (size_t)t * SD + idx;
    v += (X[o] - m) * r;
    float sp = (v >= 1.f) ? 1.f : 0.f;
    X[o] = sp;
    v = (sp != 0.f) ? 0.f : v;
  }
}

// ---------------------------------------------------------------------------
// 6) QK = sum_s (Qh * Kh) : block per (t,h); thread = (chunk, dd)
// ---------------------------------------------------------------------------
__global__ void qk_reduce_kernel(const float* __restrict__ Qs,
                                 const float* __restrict__ Ks,
                                 float* __restrict__ qk) {
  const int t = blockIdx.x / NH;
  const int h = blockIdx.x - t * NH;
  const int dd = threadIdx.x & 63;
  const int chunk = threadIdx.x >> 6;
  const size_t base = (size_t)t * SD + h * HD + dd;
  float sum = 0.f;
  for (int s = chunk * 512; s < chunk * 512 + 512; ++s) {
    size_t o = base + (size_t)s * DD;
    sum += Qs[o] * Ks[o];
  }
  __shared__ float red[256];
  red[threadIdx.x] = sum;
  __syncthreads();
  if (threadIdx.x < 64) {
    float tot = red[dd] + red[dd + 64] + red[dd + 128] + red[dd + 192];
    qk[t * DD + h * HD + dd] = tot;
  }
}

// 7) IF node on QK (T, 768)
__global__ void qk_ifnode_kernel(const float* __restrict__ qk, float* __restrict__ qks) {
  int idx = blockIdx.x * 256 + threadIdx.x;
  if (idx >= DD) return;
  float v = 0.f;
  for (int t = 0; t < TT; ++t) {
    v += qk[t * DD + idx];
    float sp = (v >= 1.f) ? 1.f : 0.f;
    qks[t * DD + idx] = sp;
    v = (sp != 0.f) ? 0.f : v;
  }
}

// 8) QKV = Vh * QKspike (broadcast over s) -> bf16 A for final GEMM (exact {0,1})
__global__ void qkv_mul_kernel(const float* __restrict__ Vs,
                               const float* __restrict__ qks,
                               bf16_t* __restrict__ A2) {
  int idx = blockIdx.x * 256 + threadIdx.x;        // over S*D
  if (idx >= SD) return;
  int d = idx % DD;
  for (int t = 0; t < TT; ++t) {
    size_t o = (size_t)t * SD + idx;
    A2[o] = (bf16_t)(Vs[o] * qks[t * DD + d]);
  }
}

// 9) Final BN normalize -> out (f32)
__global__ void bn_apply_kernel(const float* __restrict__ Y,
                                const float* __restrict__ mean,
                                const float* __restrict__ rstd,
                                float* __restrict__ out) {
  size_t idx = (size_t)blockIdx.x * 256 + threadIdx.x;  // over T*S*D
  if (idx >= (size_t)TSD) return;
  int s = (int)((idx / DD) % SS);
  out[idx] = (Y[idx] - mean[s]) * rstd[s];
}

// ---------------------------------------------------------------------------
// Launch. Workspace (~89 MB):
//   wqT/wkT/wvT/woT bf16 : 4 * 1,179,648
//   tmp  bf16 (reused as A2) : 12,582,912
//   Q f32 (reused as Y)      : 25,165,824
//   K f32, V f32             : 2 * 25,165,824
//   qk, qks                  : 2 * 12,288
//   stats (8 vecs of 2048)   : 65,536
// ---------------------------------------------------------------------------
extern "C" void kernel_launch(void* const* d_in, const int* in_sizes, int n_in,
                              void* d_out, int out_size, void* d_ws, size_t ws_size,
                              hipStream_t stream) {
  const float* x  = (const float*)d_in[0];
  const float* wq = (const float*)d_in[1];
  const float* wk = (const float*)d_in[2];
  const float* wv = (const float*)d_in[3];
  const float* wo = (const float*)d_in[4];
  float* out = (float*)d_out;
  char* ws = (char*)d_ws;

  size_t off = 0;
  const size_t WSZ = (size_t)DD * DD * sizeof(bf16_t);
  bf16_t* wqT = (bf16_t*)(ws + off); off += WSZ;
  bf16_t* wkT = (bf16_t*)(ws + off); off += WSZ;
  bf16_t* wvT = (bf16_t*)(ws + off); off += WSZ;
  bf16_t* woT = (bf16_t*)(ws + off); off += WSZ;
  bf16_t* tmp = (bf16_t*)(ws + off); off += (size_t)TSD * sizeof(bf16_t);  // also A2
  float*  Q   = (float*)(ws + off);  off += (size_t)TSD * sizeof(float);   // also Y
  float*  K   = (float*)(ws + off);  off += (size_t)TSD * sizeof(float);
  float*  V   = (float*)(ws + off);  off += (size_t)TSD * sizeof(float);
  float*  qk  = (float*)(ws + off);  off += TT * DD * sizeof(float);
  float*  qks = (float*)(ws + off);  off += TT * DD * sizeof(float);
  float*  meanQ = (float*)(ws + off); off += SS * sizeof(float);
  float*  rstdQ = (float*)(ws + off); off += SS * sizeof(float);
  float*  meanK = (float*)(ws + off); off += SS * sizeof(float);
  float*  rstdK = (float*)(ws + off); off += SS * sizeof(float);
  float*  meanV = (float*)(ws + off); off += SS * sizeof(float);
  float*  rstdV = (float*)(ws + off); off += SS * sizeof(float);
  float*  meanY = (float*)(ws + off); off += SS * sizeof(float);
  float*  rstdY = (float*)(ws + off); off += SS * sizeof(float);
  bf16_t* A2 = tmp;   // tmp is dead once the 3 QKV GEMMs finish
  float*  Y  = Q;     // Q spikes are dead once qk_reduce finishes

  const int WBLK = (DD * DD + 255) / 256;   // 2304
  const int EBLK = (SD + 255) / 256;        // 6144
  dim3 gemmGrid(8192 / TM, DD / TN);        // (64, 6)

  // weights -> bf16, transposed
  convert_w_kernel<<<WBLK, 256, 0, stream>>>(wq, wqT);
  convert_w_kernel<<<WBLK, 256, 0, stream>>>(wk, wkT);
  convert_w_kernel<<<WBLK, 256, 0, stream>>>(wv, wvT);
  convert_w_kernel<<<WBLK, 256, 0, stream>>>(wo, woT);

  // init spikes
  ifnode0_kernel<<<EBLK, 256, 0, stream>>>(x, tmp);

  // Q/K/V GEMMs (A shared)
  gemm_bf16_wmma_kernel<<<gemmGrid, 256, 0, stream>>>(tmp, wqT, Q, TT * SS, DD, DD);
  gemm_bf16_wmma_kernel<<<gemmGrid, 256, 0, stream>>>(tmp, wkT, K, TT * SS, DD, DD);
  gemm_bf16_wmma_kernel<<<gemmGrid, 256, 0, stream>>>(tmp, wvT, V, TT * SS, DD, DD);

  // BN + IF per tensor (in place -> spikes)
  bn_stats_kernel<<<SS, 256, 0, stream>>>(Q, meanQ, rstdQ);
  bn_stats_kernel<<<SS, 256, 0, stream>>>(K, meanK, rstdK);
  bn_stats_kernel<<<SS, 256, 0, stream>>>(V, meanV, rstdV);
  bn_ifnode_kernel<<<EBLK, 256, 0, stream>>>(Q, meanQ, rstdQ);
  bn_ifnode_kernel<<<EBLK, 256, 0, stream>>>(K, meanK, rstdK);
  bn_ifnode_kernel<<<EBLK, 256, 0, stream>>>(V, meanV, rstdV);

  // talking-heads reduction + IF
  qk_reduce_kernel<<<TT * NH, 256, 0, stream>>>(Q, K, qk);
  qk_ifnode_kernel<<<(DD + 255) / 256, 256, 0, stream>>>(qk, qks);

  // broadcast multiply -> bf16 A2, final GEMM, final BN
  qkv_mul_kernel<<<EBLK, 256, 0, stream>>>(V, qks, A2);
  gemm_bf16_wmma_kernel<<<gemmGrid, 256, 0, stream>>>(A2, woT, Y, TT * SS, DD, DD);
  bn_stats_kernel<<<SS, 256, 0, stream>>>(Y, meanY, rstdY);
  bn_apply_kernel<<<(TSD + 255) / 256, 256, 0, stream>>>(Y, meanY, rstdY, out);
}
